// QSelfAttention_15066745274958
// MI455X (gfx1250) — matrix operand order repaired and checked by reading, not compiled, not run on hardware
//
#include <hip/hip_runtime.h>
#include <hip/hip_bf16.h>

// ---------------------------------------------------------------------------
// QSelfAttention for MI455X (gfx1250): bf16 WMMA everywhere, flash-style
// softmax, TDM (tensor_load_to_lds) double-buffered staging for attention K/V
// and for the output-projection O tiles.  Intermediates bf16 so the ~36MB
// working set lives in the 192MB L2.  Workspace layout (bytes):
//   [0,   2MB)  Q   bf16 [B][N][KC]   (n-major)
//   [2MB, 4MB)  Kt  bf16 [B][N][KC]   (k stored transposed, n-major)
//   [4MB,20MB)  Vt  bf16 [B][C][N]    (channel-major)
//   [20MB,36MB) O   bf16 [B][N][C]    (attention output, n-major)
// ---------------------------------------------------------------------------

typedef __bf16 bf16;
typedef __attribute__((ext_vector_type(2)))  __bf16 bf16x2;
typedef __attribute__((ext_vector_type(8)))  __bf16 bf16x8;
typedef __attribute__((ext_vector_type(16))) __bf16 bf16x16;
typedef __attribute__((ext_vector_type(8)))  float  f32x8;
typedef __attribute__((ext_vector_type(4)))  unsigned int u32x4;
typedef __attribute__((ext_vector_type(4)))  int i32x4;
typedef __attribute__((ext_vector_type(8)))  int i32x8;

#define WMMA_BF16(A, B, Cacc) \
  __builtin_amdgcn_wmma_f32_16x16x32_bf16(false, (A), false, (B), (short)0, (Cacc), false, false)

static __device__ __forceinline__ bf16x16 bfcat(bf16x8 lo, bf16x8 hi) {
  return __builtin_shufflevector(lo, hi, 0,1,2,3,4,5,6,7,8,9,10,11,12,13,14,15);
}

constexpr int CC  = 512;   // channels
constexpr int NN  = 1024;  // H*W
constexpr int KCH = 64;    // key channels

// ---------------------------------------------------------------------------
// Tensor Data Mover: 2D tile (bf16 elements) global -> LDS.
// D# layout per CDNA5 ISA.
// ---------------------------------------------------------------------------
static __device__ __forceinline__ void tdm_load_2d_bf16(
    unsigned lds_addr, const void* gaddr,
    unsigned tile_d0, unsigned tile_d1,
    unsigned tensor_d0, unsigned tensor_d1, unsigned stride0)
{
  unsigned long long ga = (unsigned long long)(uintptr_t)gaddr;
  u32x4 g0;
  g0[0] = 1u;                                            // count=1, user D#
  g0[1] = lds_addr;                                      // LDS byte offset
  g0[2] = (unsigned)(ga & 0xFFFFFFFFu);                  // global_addr lo
  g0[3] = (unsigned)((ga >> 32) & 0x01FFFFFFu) | (2u << 30);  // hi | type=2
  i32x8 g1;
  g1[0] = (int)(1u << 16);                               // data_size=1 (2B)
  g1[1] = (int)((tensor_d0 & 0xFFFFu) << 16);            // tensor_dim0 lo16
  g1[2] = (int)((tensor_d0 >> 16) | ((tensor_d1 & 0xFFFFu) << 16));
  g1[3] = (int)((tensor_d1 >> 16) | (tile_d0 << 16));    // tile_dim0
  g1[4] = (int)(tile_d1);                                // tile_dim1, dim2=0
  g1[5] = (int)stride0;                                  // dim0 stride lo32
  g1[6] = 0;
  g1[7] = 0;
  i32x4 z4 = {0, 0, 0, 0};
#if __clang_major__ >= 23
  i32x8 z8 = {0, 0, 0, 0, 0, 0, 0, 0};
  __builtin_amdgcn_tensor_load_to_lds(g0, g1, z4, z4, z8, 0);
#else
  __builtin_amdgcn_tensor_load_to_lds(g0, g1, z4, z4, 0);
#endif
}

// ---------------------------------------------------------------------------
// Kernel 1: fused QKV projection.  D[n][o] = sum_c W[o][c]*X[c][n],
// o in [0,640) = {Wq rows, Wk rows, Wv rows}.  LDS-staged, fp32 -> bf16.
// Workgroup tile: 128 n x 128 o, 8 waves (each wave: 16 n-rows, 8 o-tiles).
// ---------------------------------------------------------------------------
__global__ __launch_bounds__(256) void qkv_kernel(
    const float* __restrict__ x,
    const float* __restrict__ Wq, const float* __restrict__ bq,
    const float* __restrict__ Wk, const float* __restrict__ bk,
    const float* __restrict__ Wv, const float* __restrict__ bv,
    bf16* __restrict__ Q, bf16* __restrict__ Kt, bf16* __restrict__ Vt)
{
  __shared__ bf16 sA[128][32];  // X^T tile: [n_local][c_local]
  __shared__ bf16 sB[128][32];  // W   tile: [o_local][c_local]

  const int b   = blockIdx.z;
  const int n0  = blockIdx.y * 128;
  const int o0  = blockIdx.x * 128;
  const int t   = threadIdx.x;
  const int w   = t >> 5;
  const int L   = t & 31;
  const int col = L & 15;
  const int hi  = L >> 4;

  const float* xb = x + (size_t)b * CC * NN;

  f32x8 acc[8] = {};

  for (int c0 = 0; c0 < CC; c0 += 32) {
    __syncthreads();
    for (int e = t; e < 32 * 128; e += 256) {       // X^T stage (transpose)
      int cl = e >> 7, nl = e & 127;
      sA[nl][cl] = (bf16)xb[(size_t)(c0 + cl) * NN + n0 + nl];
    }
    for (int e = t; e < 128 * 32; e += 256) {       // W stage (fp32 -> bf16)
      int ol = e >> 5, cl = e & 31;
      int o  = o0 + ol;
      float wv;
      if (o < 64)       wv = Wq[o * CC + c0 + cl];
      else if (o < 128) wv = Wk[(o - 64) * CC + c0 + cl];
      else              wv = Wv[(size_t)(o - 128) * CC + c0 + cl];
      sB[ol][cl] = (bf16)wv;
    }
    __syncthreads();

    const int m = w * 16 + col;
    bf16x16 a = bfcat(*(const bf16x8*)&sA[m][hi * 16],
                      *(const bf16x8*)&sA[m][hi * 16 + 8]);
    bf16x16 bm[8];
#pragma unroll
    for (int ot = 0; ot < 8; ++ot) {
      const int o = ot * 16 + col;
      bm[ot] = bfcat(*(const bf16x8*)&sB[o][hi * 16],
                     *(const bf16x8*)&sB[o][hi * 16 + 8]);
    }
#pragma unroll
    for (int ot = 0; ot < 8; ++ot) acc[ot] = WMMA_BF16(a, bm[ot], acc[ot]);
  }

  const int nbase = n0 + w * 16 + hi * 8;
#pragma unroll
  for (int ot = 0; ot < 8; ++ot) {
    const int o = o0 + ot * 16 + col;
    if (o < 64) {
      const float bias = bq[o];
      bf16* q = Q + ((size_t)b * NN + nbase) * KCH + o;
#pragma unroll
      for (int r = 0; r < 8; ++r) q[(size_t)r * KCH] = (bf16)(acc[ot][r] + bias);
    } else if (o < 128) {
      const float bias = bk[o - 64];
      bf16* kk = Kt + ((size_t)b * NN + nbase) * KCH + (o - 64);
#pragma unroll
      for (int r = 0; r < 8; ++r) kk[(size_t)r * KCH] = (bf16)(acc[ot][r] + bias);
    } else {
      const float bias = bv[o - 128];
      bf16* vv = Vt + ((size_t)b * CC + (o - 128)) * NN + nbase;
#pragma unroll
      for (int r = 0; r < 8; r += 2) {
        bf16x2 p = { (bf16)(acc[ot][r] + bias), (bf16)(acc[ot][r + 1] + bias) };
        *(bf16x2*)&vv[r] = p;
      }
    }
  }
}

// ---------------------------------------------------------------------------
// Kernel 2: flash attention with TDM double-buffered K/V staging.
// WG = 8 waves = 2 query tiles (16 rows) x 4 C-slices (128 v-channels).
// Wave 0 issues tensor_load_to_lds for chunk i+2 while everyone computes on
// chunk i; sync = s_wait_tensorcnt + workgroup barrier.
// ---------------------------------------------------------------------------
__global__ __launch_bounds__(256) void attn_kernel(
    const bf16* __restrict__ Q, const bf16* __restrict__ Kt,
    const bf16* __restrict__ Vt, bf16* __restrict__ O)
{
  __shared__ bf16 sK[2][32][64];    // key chunk   [key][kc]      2 x 4KB
  __shared__ bf16 sV[2][512][32];   // value chunk [vchan][key]   2 x 32KB
  __shared__ bf16 sP[8][16][32];    // per-wave P  [row][key]     8KB

  const int b   = blockIdx.y;
  const int nb  = blockIdx.x;
  const int t   = threadIdx.x;
  const int w   = t >> 5;
  const int L   = t & 31;
  const int col = L & 15;
  const int hi  = L >> 4;
  const int qt  = w >> 2;
  const int cs  = (w & 3) * 128;
  const int n0  = nb * 32 + qt * 16;

  const bf16* Ktb = Kt + (size_t)b * NN * KCH;
  const bf16* Vtb = Vt + (size_t)b * CC * NN;

  // Q A-operands (KC = 64 -> two K=32 steps), register-resident for the loop
  const bf16* qrow = Q + ((size_t)b * NN + n0 + col) * KCH;
  bf16x16 aq0 = bfcat(*(const bf16x8*)&qrow[hi * 16],
                      *(const bf16x8*)&qrow[hi * 16 + 8]);
  bf16x16 aq1 = bfcat(*(const bf16x8*)&qrow[32 + hi * 16],
                      *(const bf16x8*)&qrow[32 + hi * 16 + 8]);

  f32x8 acc[8] = {};
  float mrow[8], lrow[8];
#pragma unroll
  for (int r = 0; r < 8; ++r) { mrow[r] = -3.0e38f; lrow[r] = 0.0f; }

  constexpr float scale = 0.125f;  // KC^-0.5
  constexpr int NCHUNK = NN / 32;  // 32 key chunks

  // prologue: chunks 0 and 1 in flight (2 TDM ops each: K tile + V tile)
  if (w == 0) {
#pragma unroll
    for (int c = 0; c < 2; ++c) {
      const int k0 = c * 32;
      tdm_load_2d_bf16((unsigned)(uintptr_t)&sK[c][0][0],
                       Ktb + (size_t)k0 * KCH,
                       /*tile*/ KCH, 32, /*tensor*/ KCH, NN, /*stride*/ KCH);
      tdm_load_2d_bf16((unsigned)(uintptr_t)&sV[c][0][0],
                       Vtb + k0,
                       /*tile*/ 32, CC, /*tensor*/ NN, CC, /*stride*/ NN);
    }
  }

  for (int it = 0; it < NCHUNK; ++it) {
    const int buf = it & 1;

    if (w == 0) {
      if (it < NCHUNK - 1) __builtin_amdgcn_s_wait_tensorcnt(2);  // chunk it done
      else                 __builtin_amdgcn_s_wait_tensorcnt(0);  // last chunk
    }
    __syncthreads();

    // S tiles for keys [k0,k0+16) and [k0+16,k0+32)
    f32x8 s0 = {}, s1 = {};
    {
      bf16x16 b00 = bfcat(*(const bf16x8*)&sK[buf][col][hi * 16],
                          *(const bf16x8*)&sK[buf][col][hi * 16 + 8]);
      bf16x16 b01 = bfcat(*(const bf16x8*)&sK[buf][col][32 + hi * 16],
                          *(const bf16x8*)&sK[buf][col][32 + hi * 16 + 8]);
      bf16x16 b10 = bfcat(*(const bf16x8*)&sK[buf][16 + col][hi * 16],
                          *(const bf16x8*)&sK[buf][16 + col][hi * 16 + 8]);
      bf16x16 b11 = bfcat(*(const bf16x8*)&sK[buf][16 + col][32 + hi * 16],
                          *(const bf16x8*)&sK[buf][16 + col][32 + hi * 16 + 8]);
      s0 = WMMA_BF16(aq0, b00, s0);
      s0 = WMMA_BF16(aq1, b01, s0);
      s1 = WMMA_BF16(aq0, b10, s1);
      s1 = WMMA_BF16(aq1, b11, s1);
    }

    // online softmax update (one row per accumulator element, 16-lane groups)
#pragma unroll
    for (int r = 0; r < 8; ++r) {
      float e0 = s0[r] * scale, e1 = s1[r] * scale;
      float cm = fmaxf(e0, e1);
#pragma unroll
      for (int d = 1; d < 16; d <<= 1) cm = fmaxf(cm, __shfl_xor(cm, d, 32));
      float mn    = fmaxf(mrow[r], cm);
      float alpha = __expf(mrow[r] - mn);
      float p0 = __expf(e0 - mn);
      float p1 = __expf(e1 - mn);
      float rs = p0 + p1;
#pragma unroll
      for (int d = 1; d < 16; d <<= 1) rs += __shfl_xor(rs, d, 32);
      lrow[r] = lrow[r] * alpha + rs;
      mrow[r] = mn;
#pragma unroll
      for (int tt = 0; tt < 8; ++tt) acc[tt][r] *= alpha;
      sP[w][r + hi * 8][col]      = (bf16)p0;   // D layout -> [row][key]
      sP[w][r + hi * 8][16 + col] = (bf16)p1;
    }
    // same-wave LDS RAW across lanes: drain DS counter before re-reading
    asm volatile("s_wait_dscnt 0" ::: "memory");

    bf16x16 ap = bfcat(*(const bf16x8*)&sP[w][col][hi * 16],
                       *(const bf16x8*)&sP[w][col][hi * 16 + 8]);
    bf16x16 bv2[8];
#pragma unroll
    for (int tt = 0; tt < 8; ++tt) {
      const int vch = cs + tt * 16 + col;
      bv2[tt] = bfcat(*(const bf16x8*)&sV[buf][vch][hi * 16],
                      *(const bf16x8*)&sV[buf][vch][hi * 16 + 8]);
    }
#pragma unroll
    for (int tt = 0; tt < 8; ++tt) acc[tt] = WMMA_BF16(ap, bv2[tt], acc[tt]);

    __syncthreads();   // everyone done reading buf before TDM overwrites it
    if (w == 0 && it + 2 < NCHUNK) {
      const int kn = (it + 2) * 32;
      tdm_load_2d_bf16((unsigned)(uintptr_t)&sK[buf][0][0],
                       Ktb + (size_t)kn * KCH, KCH, 32, KCH, NN, KCH);
      tdm_load_2d_bf16((unsigned)(uintptr_t)&sV[buf][0][0],
                       Vtb + kn, 32, CC, NN, CC, NN);
    }
  }

  // normalize and store O (n-major bf16)
#pragma unroll
  for (int r = 0; r < 8; ++r) {
    const float inv = 1.0f / lrow[r];
#pragma unroll
    for (int tt = 0; tt < 8; ++tt) {
      const int vch = cs + tt * 16 + col;
      O[((size_t)b * NN + n0 + hi * 8 + r) * CC + vch] = (bf16)(acc[tt][r] * inv);
    }
  }
}

// ---------------------------------------------------------------------------
// Kernel 3: y = gamma * (Wo . O + bo) + x, written channel-major fp32.
// Workgroup tile: 128 n x 128 co.  O tiles staged by TDM (double-buffered);
// Wo tiles staged by VALU (needs fp32->bf16) into the other buffer during
// the compute phase of the previous step.
// ---------------------------------------------------------------------------
__global__ __launch_bounds__(256) void out_kernel(
    const bf16* __restrict__ O, const float* __restrict__ Wo,
    const float* __restrict__ bo, const float* __restrict__ x,
    const float* __restrict__ gamma, float* __restrict__ y)
{
  __shared__ bf16 sA[2][128][32];  // O  tile [n][v]   2 x 8KB (TDM)
  __shared__ bf16 sB[2][128][32];  // Wo tile [co][v]  2 x 8KB (VALU)

  const int b   = blockIdx.z;
  const int n0  = blockIdx.y * 128;
  const int o0  = blockIdx.x * 128;
  const int t   = threadIdx.x;
  const int w   = t >> 5;
  const int L   = t & 31;
  const int col = L & 15;
  const int hi  = L >> 4;

  const bf16* Ob = O + ((size_t)b * NN + n0) * CC;   // tile base row

  f32x8 acc[8] = {};

  // prologue: Wo step0 via VALU, O steps 0 and 1 via TDM
  for (int e = t; e < 128 * 32; e += 256) {
    int ol = e >> 5, cl = e & 31;
    sB[0][ol][cl] = (bf16)Wo[(size_t)(o0 + ol) * CC + cl];
  }
  if (w == 0) {
    tdm_load_2d_bf16((unsigned)(uintptr_t)&sA[0][0][0], Ob +  0,
                     /*tile*/ 32, 128, /*tensor*/ CC, NN, /*stride*/ CC);
    tdm_load_2d_bf16((unsigned)(uintptr_t)&sA[1][0][0], Ob + 32,
                     /*tile*/ 32, 128, /*tensor*/ CC, NN, /*stride*/ CC);
  }

  constexpr int NSTEP = CC / 32;  // 16 k-steps
  for (int i = 0; i < NSTEP; ++i) {
    const int buf = i & 1;
    if (w == 0) {
      if (i < NSTEP - 1) __builtin_amdgcn_s_wait_tensorcnt(1);  // step i landed
      else               __builtin_amdgcn_s_wait_tensorcnt(0);
    }
    __syncthreads();

    const int m = w * 16 + col;
    bf16x16 a = bfcat(*(const bf16x8*)&sA[buf][m][hi * 16],
                      *(const bf16x8*)&sA[buf][m][hi * 16 + 8]);
    bf16x16 bm[8];
#pragma unroll
    for (int ot = 0; ot < 8; ++ot) {
      const int o = ot * 16 + col;
      bm[ot] = bfcat(*(const bf16x8*)&sB[buf][o][hi * 16],
                     *(const bf16x8*)&sB[buf][o][hi * 16 + 8]);
    }
#pragma unroll
    for (int ot = 0; ot < 8; ++ot) acc[ot] = WMMA_BF16(a, bm[ot], acc[ot]);

    // stage Wo for step i+1 into the other buffer while this step computes
    if (i + 1 < NSTEP) {
      const int v1 = (i + 1) * 32;
      for (int e = t; e < 128 * 32; e += 256) {
        int ol = e >> 5, cl = e & 31;
        sB[buf ^ 1][ol][cl] = (bf16)Wo[(size_t)(o0 + ol) * CC + v1 + cl];
      }
    }
    __syncthreads();   // all reads of sA[buf]/writes of sB[buf^1] done
    if (w == 0 && i + 2 < NSTEP) {
      tdm_load_2d_bf16((unsigned)(uintptr_t)&sA[buf][0][0],
                       Ob + (i + 2) * 32, 32, 128, CC, NN, CC);
    }
  }

  const float g     = gamma[0];
  const int   nbase = n0 + w * 16 + hi * 8;
#pragma unroll
  for (int ot = 0; ot < 8; ++ot) {
    const int   co   = o0 + ot * 16 + col;
    const float bias = bo[co];
    const float* xr = x + ((size_t)b * CC + co) * NN + nbase;
    float*       yr = y + ((size_t)b * CC + co) * NN + nbase;
#pragma unroll
    for (int r = 0; r < 8; r += 2) {
      float2 xv = *(const float2*)&xr[r];
      float2 o2;
      o2.x = g * (acc[ot][r]     + bias) + xv.x;
      o2.y = g * (acc[ot][r + 1] + bias) + xv.y;
      *(float2*)&yr[r] = o2;
    }
  }
}

// ---------------------------------------------------------------------------
extern "C" void kernel_launch(void* const* d_in, const int* in_sizes, int n_in,
                              void* d_out, int out_size, void* d_ws, size_t ws_size,
                              hipStream_t stream) {
  const float* x     = (const float*)d_in[0];
  const float* Wq    = (const float*)d_in[1];
  const float* bq    = (const float*)d_in[2];
  const float* Wk    = (const float*)d_in[3];
  const float* bk    = (const float*)d_in[4];
  const float* Wv    = (const float*)d_in[5];
  const float* bv    = (const float*)d_in[6];
  const float* Wo    = (const float*)d_in[7];
  const float* bo    = (const float*)d_in[8];
  const float* gamma = (const float*)d_in[9];

  char* ws = (char*)d_ws;                       // needs >= 36MB
  bf16* Q  = (bf16*)(ws);
  bf16* Kt = (bf16*)(ws + (size_t)2  * 1024 * 1024);
  bf16* Vt = (bf16*)(ws + (size_t)4  * 1024 * 1024);
  bf16* O  = (bf16*)(ws + (size_t)20 * 1024 * 1024);

  qkv_kernel <<<dim3(5, 8, 16), 256, 0, stream>>>(x, Wq, bq, Wk, bk, Wv, bv, Q, Kt, Vt);
  attn_kernel<<<dim3(32, 16),   256, 0, stream>>>(Q, Kt, Vt, O);
  out_kernel <<<dim3(4, 8, 16), 256, 0, stream>>>(O, Wo, bo, x, gamma, (float*)d_out);
}
